// CrossGraphAttention_50337016709459
// MI455X (gfx1250) — matrix operand order repaired and checked by reading, not compiled, not run on hardware
//
#include <hip/hip_runtime.h>
#include <hip/hip_bf16.h>

typedef __attribute__((ext_vector_type(16))) __bf16 v16bf;
typedef __attribute__((ext_vector_type(8)))  __bf16 v8bf;
typedef __attribute__((ext_vector_type(8)))  float  v8f;

#define Bc 8
#define Nc 2048
#define Dc 256

__device__ __forceinline__ v16bf cat8(v8bf lo, v8bf hi) {
  return __builtin_shufflevector(lo, hi, 0,1,2,3,4,5,6,7,8,9,10,11,12,13,14,15);
}
__device__ __forceinline__ v8bf cvt8(v8f x) {
  return __builtin_convertvector(x, v8bf);
}
__device__ __forceinline__ float rmax16(float v) {
  v = fmaxf(v, __shfl_xor(v, 1, 32));
  v = fmaxf(v, __shfl_xor(v, 2, 32));
  v = fmaxf(v, __shfl_xor(v, 4, 32));
  v = fmaxf(v, __shfl_xor(v, 8, 32));
  return v;
}
__device__ __forceinline__ float rsum16(float v) {
  v += __shfl_xor(v, 1, 32);
  v += __shfl_xor(v, 2, 32);
  v += __shfl_xor(v, 4, 32);
  v += __shfl_xor(v, 8, 32);
  return v;
}
__device__ __forceinline__ v8f wmma_bf16(v16bf a, v16bf b, v8f c) {
  return __builtin_amdgcn_wmma_f32_16x16x32_bf16(false, a, false, b, (short)0, c, false, false);
}
// 16-byte global -> LDS async copy (ASYNCcnt tracked)
__device__ __forceinline__ void async_cp16(unsigned lds_off, unsigned long long gaddr) {
  asm volatile("global_load_async_to_lds_b128 %0, %1, off"
               :: "v"(lds_off), "v"(gaddr) : "memory");
}
__device__ __forceinline__ unsigned lds_lo32(const void* p) {
  return (unsigned)(unsigned long long)p;  // flat LDS addr low 32 bits = group-relative offset
}

// ---- prep: bf16 copy of x2, plus d-major transposes of x1/x2 ----
__global__ void k_prep(const float* __restrict__ x1, const float* __restrict__ x2,
                       __bf16* __restrict__ x2b, __bf16* __restrict__ x2t,
                       __bf16* __restrict__ x1t) {
  long i = (long)blockIdx.x * blockDim.x + threadIdx.x;
  const long total = (long)Bc * Nc * Dc;
  if (i >= total) return;
  int d = (int)(i % Dc);
  long r = i / Dc;
  int n = (int)(r % Nc);
  int b = (int)(r / Nc);
  float v2 = x2[i];
  x2b[i] = (__bf16)v2;
  x2t[((long)b * Dc + d) * Nc + n] = (__bf16)v2;
  x1t[((long)b * Dc + d) * Nc + n] = (__bf16)x1[i];
}

// ---- q = x1 @ W^T + bias, output bf16 ----
__global__ void k_qgemm(const float* __restrict__ x1, const float* __restrict__ W,
                        const float* __restrict__ bias, __bf16* __restrict__ qb) {
  int wave = (blockIdx.x * blockDim.x + threadIdx.x) >> 5;
  int lane = threadIdx.x & 31;
  int ln = lane & 15, hi = lane >> 4;
  int colTile = wave & 15;
  int rowTile = (wave >> 4) & 127;
  int b = wave >> 11;
  const float* A = x1 + ((long)b * Nc + rowTile * 16) * Dc;
  int e = colTile * 16 + ln;
  const float* Brow = W + (long)e * Dc;
  v8f acc = {};
  #pragma unroll
  for (int k0 = 0; k0 < Dc; k0 += 32) {
    v8f alo = *(const v8f*)(A + ln * Dc + k0 + hi * 8);
    v8f ahi = *(const v8f*)(A + ln * Dc + k0 + 16 + hi * 8);
    v8f blo = *(const v8f*)(Brow + k0 + hi * 8);
    v8f bhi = *(const v8f*)(Brow + k0 + 16 + hi * 8);
    v16bf a  = cat8(cvt8(alo), cvt8(ahi));
    v16bf bb = cat8(cvt8(blo), cvt8(bhi));
    acc = wmma_bf16(a, bb, acc);
  }
  float bv = bias[e];
  __bf16* out = qb + ((long)b * Nc + rowTile * 16) * Dc + colTile * 16 + ln;
  #pragma unroll
  for (int v = 0; v < 8; ++v)
    out[(long)(v + 8 * hi) * Dc] = (__bf16)(acc[v] + bv);
}

// ---- per-row softmax stats: rowmax m, inv sum-exp ----
__global__ void k_stats(const __bf16* __restrict__ qb, const __bf16* __restrict__ x2b,
                        float* __restrict__ Mst, float* __restrict__ ILst) {
  int wave = (blockIdx.x * blockDim.x + threadIdx.x) >> 5;
  int lane = threadIdx.x & 31;
  int ln = lane & 15, hi = lane >> 4;
  int rowTile = wave & 127;
  int b = wave >> 7;
  const __bf16* Q = qb + ((long)b * Nc + rowTile * 16) * Dc;
  v16bf aF[8];
  #pragma unroll
  for (int kk = 0; kk < 8; ++kk) {
    const __bf16* p = Q + ln * Dc + kk * 32 + hi * 8;
    aF[kk] = cat8(*(const v8bf*)p, *(const v8bf*)(p + 16));
  }
  float mr[8], lr[8];
  #pragma unroll
  for (int v = 0; v < 8; ++v) { mr[v] = -1e30f; lr[v] = 0.f; }
  const __bf16* X2 = x2b + (long)b * Nc * Dc;
  for (int n0 = 0; n0 < Nc; n0 += 16) {
    const __bf16* Bp = X2 + (long)(n0 + ln) * Dc;
    v8f acc = {};
    #pragma unroll
    for (int kk = 0; kk < 8; ++kk) {
      const __bf16* p = Bp + kk * 32 + hi * 8;
      v16bf bb = cat8(*(const v8bf*)p, *(const v8bf*)(p + 16));
      acc = wmma_bf16(aF[kk], bb, acc);
    }
    #pragma unroll
    for (int v = 0; v < 8; ++v) {
      float s = acc[v];
      float nm = fmaxf(mr[v], rmax16(s));
      float ps = rsum16(__expf(s - nm));
      lr[v] = lr[v] * __expf(mr[v] - nm) + ps;
      mr[v] = nm;
    }
  }
  if (ln == 0) {
    #pragma unroll
    for (int v = 0; v < 8; ++v) {
      long r = (long)b * Nc + rowTile * 16 + v + 8 * hi;
      Mst[r] = mr[v];
      ILst[r] = 1.0f / lr[v];
    }
  }
}

// Stage 16KB contiguous slab + 256x32 transposed chunk into LDS (per block).
// s0: 32 rows x 256 bf16 (row-major slab, contiguous in global)
// s1: 256 rows x 32 bf16 (row d of transposed tensor, 64B per row)
__device__ __forceinline__ void stage_slabs(const __bf16* slab_g, const __bf16* tr_g_base,
                                            long tr_row_stride_elems, int n0,
                                            __bf16* s0dst, __bf16* s1dst, int tid) {
  const char* g0 = (const char*)slab_g;
  unsigned l0 = lds_lo32(s0dst);
  #pragma unroll
  for (int i = 0; i < 8; ++i) {
    unsigned off = (unsigned)(i * 128 + tid) * 16u;
    async_cp16(l0 + off, (unsigned long long)(g0 + off));
  }
  const char* g1 = (const char*)(tr_g_base + n0);
  unsigned l1 = lds_lo32(s1dst);
  #pragma unroll
  for (int i = 0; i < 8; ++i) {
    int c = i * 128 + tid;          // 0..1023 chunks of 16B
    int d = c >> 2;
    int off = (c & 3) * 16;
    async_cp16(l1 + (unsigned)(d * 64 + off),
               (unsigned long long)(g1 + (long)d * tr_row_stride_elems * 2 + off));
  }
}

// ---- x1_updated = softmax(S) @ x2 ; wave owns 16 rows x 256 cols ----
__global__ void k_x1u(const __bf16* __restrict__ qb, const __bf16* __restrict__ x2b,
                      const __bf16* __restrict__ x2t,
                      const float* __restrict__ Mst, const float* __restrict__ ILst,
                      float* __restrict__ out1) {
  __shared__ __align__(16) __bf16 s0[2][32 * 256];   // x2b rows slab
  __shared__ __align__(16) __bf16 s1[2][256 * 32];   // x2t chunk
  __shared__ __align__(16) __bf16 plds[4][16 * 32];
  int tid = threadIdx.x;
  int wid = tid >> 5;
  int wave = (blockIdx.x * blockDim.x + tid) >> 5;
  int lane = tid & 31;
  int ln = lane & 15, hi = lane >> 4;
  int rowTile = wave & 127;
  int b = wave >> 7;
  const __bf16* Q = qb + ((long)b * Nc + rowTile * 16) * Dc;
  v16bf aF[8];
  #pragma unroll
  for (int kk = 0; kk < 8; ++kk) {
    const __bf16* p = Q + ln * Dc + kk * 32 + hi * 8;
    aF[kk] = cat8(*(const v8bf*)p, *(const v8bf*)(p + 16));
  }
  float mr[8], il[8];
  #pragma unroll
  for (int v = 0; v < 8; ++v) {
    long r = (long)b * Nc + rowTile * 16 + v + 8 * hi;
    mr[v] = Mst[r];
    il[v] = ILst[r];
  }
  v8f acc[16];
  #pragma unroll
  for (int dt = 0; dt < 16; ++dt) acc[dt] = (v8f){};
  const __bf16* X2  = x2b + (long)b * Nc * Dc;
  const __bf16* X2T = x2t + (long)b * Dc * Nc;
  __bf16* P = plds[wid];

  stage_slabs(X2, X2T, Nc, 0, s0[0], s1[0], tid);  // prologue: buffer 0
  for (int n0 = 0; n0 < Nc; n0 += 32) {
    int buf = (n0 >> 5) & 1;
    if (n0 + 32 < Nc) {
      stage_slabs(X2 + (long)(n0 + 32) * Dc, X2T, Nc, n0 + 32, s0[buf ^ 1], s1[buf ^ 1], tid);
      asm volatile("s_wait_asynccnt 16" ::: "memory");
    } else {
      asm volatile("s_wait_asynccnt 0" ::: "memory");
    }
    __syncthreads();
    #pragma unroll
    for (int t = 0; t < 2; ++t) {
      v8f s = {};
      #pragma unroll
      for (int kk = 0; kk < 8; ++kk) {
        const __bf16* p = &s0[buf][(t * 16 + ln) * 256 + kk * 32 + hi * 8];
        v16bf bb = cat8(*(const v8bf*)p, *(const v8bf*)(p + 16));
        s = wmma_bf16(aF[kk], bb, s);
      }
      #pragma unroll
      for (int v = 0; v < 8; ++v) {
        float pv = __expf(s[v] - mr[v]) * il[v];
        P[(v + 8 * hi) * 32 + t * 16 + ln] = (__bf16)pv;  // P[m][n-offset]
      }
    }
    asm volatile("s_wait_dscnt 0" ::: "memory");
    const __bf16* pr = P + ln * 32 + hi * 8;
    v16bf aP = cat8(*(const v8bf*)pr, *(const v8bf*)(pr + 16));
    #pragma unroll
    for (int dt = 0; dt < 16; ++dt) {
      const __bf16* bp = &s1[buf][(dt * 16 + ln) * 32 + hi * 8];
      v16bf bb = cat8(*(const v8bf*)bp, *(const v8bf*)(bp + 16));
      acc[dt] = wmma_bf16(aP, bb, acc[dt]);
    }
    __syncthreads();
  }
  float* O = out1 + ((long)b * Nc + rowTile * 16) * Dc;
  #pragma unroll
  for (int dt = 0; dt < 16; ++dt)
    #pragma unroll
    for (int v = 0; v < 8; ++v)
      O[(long)(v + 8 * hi) * Dc + dt * 16 + ln] = acc[dt][v];
}

// ---- x2_updated = softmax(S)^T @ x1 ; wave owns 16 n-rows x 256 cols ----
__global__ void k_x2u(const __bf16* __restrict__ qb, const __bf16* __restrict__ x2b,
                      const __bf16* __restrict__ x1t,
                      const float* __restrict__ Mst, const float* __restrict__ ILst,
                      float* __restrict__ out2) {
  __shared__ __align__(16) __bf16 s0[2][32 * 256];   // qb rows slab
  __shared__ __align__(16) __bf16 s1[2][256 * 32];   // x1t chunk
  __shared__ __align__(16) __bf16 plds[4][16 * 32];
  int tid = threadIdx.x;
  int wid = tid >> 5;
  int wave = (blockIdx.x * blockDim.x + tid) >> 5;
  int lane = tid & 31;
  int ln = lane & 15, hi = lane >> 4;
  int nTile = wave & 127;
  int b = wave >> 7;
  // Preload B fragments of x2^T for the score GEMM (column n fixed per lane)
  const __bf16* X2 = x2b + ((long)b * Nc + nTile * 16 + ln) * Dc;
  v16bf bF[8];
  #pragma unroll
  for (int kk = 0; kk < 8; ++kk) {
    const __bf16* p = X2 + kk * 32 + hi * 8;
    bF[kk] = cat8(*(const v8bf*)p, *(const v8bf*)(p + 16));
  }
  v8f acc[16];
  #pragma unroll
  for (int dt = 0; dt < 16; ++dt) acc[dt] = (v8f){};
  const __bf16* Qb  = qb + (long)b * Nc * Dc;
  const __bf16* X1T = x1t + (long)b * Dc * Nc;
  const float* Mb  = Mst + (long)b * Nc;
  const float* ILb = ILst + (long)b * Nc;
  __bf16* P = plds[wid];  // P^T layout: [n 16][m 32]

  stage_slabs(Qb, X1T, Nc, 0, s0[0], s1[0], tid);  // prologue: buffer 0
  for (int m0 = 0; m0 < Nc; m0 += 32) {
    int buf = (m0 >> 5) & 1;
    if (m0 + 32 < Nc) {
      stage_slabs(Qb + (long)(m0 + 32) * Dc, X1T, Nc, m0 + 32, s0[buf ^ 1], s1[buf ^ 1], tid);
      asm volatile("s_wait_asynccnt 16" ::: "memory");
    } else {
      asm volatile("s_wait_asynccnt 0" ::: "memory");
    }
    __syncthreads();
    #pragma unroll
    for (int t = 0; t < 2; ++t) {
      v8f s = {};
      #pragma unroll
      for (int kk = 0; kk < 8; ++kk) {
        const __bf16* p = &s0[buf][(t * 16 + ln) * 256 + kk * 32 + hi * 8];
        v16bf aa = cat8(*(const v8bf*)p, *(const v8bf*)(p + 16));
        s = wmma_bf16(aa, bF[kk], s);
      }
      #pragma unroll
      for (int v = 0; v < 8; ++v) {
        int mrow = m0 + t * 16 + v + 8 * hi;
        float pv = __expf(s[v] - Mb[mrow]) * ILb[mrow];
        P[ln * 32 + t * 16 + v + 8 * hi] = (__bf16)pv;  // PT[n=ln][m-offset]
      }
    }
    asm volatile("s_wait_dscnt 0" ::: "memory");
    const __bf16* pr = P + ln * 32 + hi * 8;
    v16bf aP = cat8(*(const v8bf*)pr, *(const v8bf*)(pr + 16));
    #pragma unroll
    for (int dt = 0; dt < 16; ++dt) {
      const __bf16* bp = &s1[buf][(dt * 16 + ln) * 32 + hi * 8];
      v16bf bb = cat8(*(const v8bf*)bp, *(const v8bf*)(bp + 16));
      acc[dt] = wmma_bf16(aP, bb, acc[dt]);
    }
    __syncthreads();
  }
  float* O = out2 + ((long)b * Nc + nTile * 16) * Dc;
  #pragma unroll
  for (int dt = 0; dt < 16; ++dt)
    #pragma unroll
    for (int v = 0; v < 8; ++v)
      O[(long)(v + 8 * hi) * Dc + dt * 16 + ln] = acc[dt][v];
}

extern "C" void kernel_launch(void* const* d_in, const int* in_sizes, int n_in,
                              void* d_out, int out_size, void* d_ws, size_t ws_size,
                              hipStream_t stream) {
  (void)in_sizes; (void)n_in; (void)out_size; (void)ws_size;
  const float* x1   = (const float*)d_in[0];
  const float* x2   = (const float*)d_in[1];
  const float* W    = (const float*)d_in[2];
  const float* bias = (const float*)d_in[3];
  float* out1 = (float*)d_out;
  float* out2 = out1 + (size_t)Bc * Nc * Dc;

  const size_t EL = (size_t)Bc * Nc * Dc;
  char* ws = (char*)d_ws;
  __bf16* qb  = (__bf16*)ws;  ws += EL * 2;
  __bf16* x2b = (__bf16*)ws;  ws += EL * 2;
  __bf16* x2t = (__bf16*)ws;  ws += EL * 2;
  __bf16* x1t = (__bf16*)ws;  ws += EL * 2;
  float* Mst  = (float*)ws;   ws += (size_t)Bc * Nc * 4;
  float* ILst = (float*)ws;

  k_prep <<<(int)((EL + 255) / 256), 256, 0, stream>>>(x1, x2, x2b, x2t, x1t);
  k_qgemm<<<4096, 128, 0, stream>>>(x1, W, bias, qb);
  k_stats<<<256, 128, 0, stream>>>(qb, x2b, Mst, ILst);
  k_x1u  <<<256, 128, 0, stream>>>(qb, x2b, x2t, Mst, ILst, out1);
  k_x2u  <<<256, 128, 0, stream>>>(qb, x2b, x1t, Mst, ILst, out2);
}
